// MessageAggregation_77472620085743
// MI455X (gfx1250) — matrix operand order repaired and checked by reading, not compiled, not run on hardware
//
#include <hip/hip_runtime.h>

typedef __attribute__((ext_vector_type(16))) _Float16 v16h;
typedef __attribute__((ext_vector_type(8)))  float    v8f;

#define KK 8
#define BB 4
#define CC 128
#define HW 9216           // 96*96
#define HW4 2304          // HW/4
#define M_ROWS 32         // K*B

// ---------------------------------------------------------------------------
// Kernel 1: pooled[k,b,c] = mean_{h,w} messages[k,b,c,h,w]
// One block per (k,b,c) plane; 256 threads x 9 float4 loads each = 9216 floats.
// ---------------------------------------------------------------------------
__global__ void pool_kernel(const float* __restrict__ msg, float* __restrict__ pooled) {
    __shared__ float sdata[256];
    const int plane = blockIdx.x;                       // 0..4095
    const float4* p = (const float4*)(msg + (size_t)plane * HW);
    float s = 0.f;
    #pragma unroll
    for (int i = 0; i < 9; ++i) {
        float4 v = p[threadIdx.x + (i << 8)];
        s += v.x + v.y + v.z + v.w;
    }
    sdata[threadIdx.x] = s;
    __syncthreads();
    #pragma unroll
    for (int off = 128; off > 0; off >>= 1) {
        if ((int)threadIdx.x < off) sdata[threadIdx.x] += sdata[threadIdx.x + off];
        __syncthreads();
    }
    if (threadIdx.x == 0) pooled[plane] = sdata[0] * (1.0f / (float)HW);
}

// ---------------------------------------------------------------------------
// Kernel 2: g[m,d] = sigmoid( sum_c pooled[m,c] * W[d,c] + b_conv[d] + b_gate[d] )
// m = k*B + b (32 rows), d = 0..127.  One workgroup, 16 waves:
//   wave -> (mtile 0..1, ntile 0..7), 4 accumulating WMMA steps over K=128.
// Fragment layouts per CDNA5 ISA 7.12.2 (16-bit A 16x32, B 32x16, f32 C/D 16x16).
// ---------------------------------------------------------------------------
__global__ void gate_kernel(const float* __restrict__ pooled,
                            const float* __restrict__ W,
                            const float* __restrict__ b_conv,
                            const float* __restrict__ b_gate,
                            float* __restrict__ g) {
    const int lane  = threadIdx.x & 31;
    const int wave  = threadIdx.x >> 5;
    const int mtile = wave >> 3;                 // 0..1
    const int ntile = wave & 7;                  // 0..7
    const int l15   = lane & 15;
    const int m     = mtile * 16 + l15;          // A row for this lane
    const int n     = ntile * 16 + l15;          // B/D column for this lane
    const int koffA = (lane < 16) ? 0 : 8;       // A: K sub-offset per half-wave
    const int koffB = (lane < 16) ? 0 : 8;       // B: K sub-offset per half-wave

    v8f acc = {};
    #pragma unroll
    for (int kk = 0; kk < 4; ++kk) {             // K chunks of 32
        const int kb = kk * 32;
        v16h a, b;
        #pragma unroll
        for (int j = 0; j < 16; ++j) {
            const int vg = j >> 1, w = j & 1;
            // A-matrix 16x32 f16 layout: VGPR0-3 -> K=koffA+{0..7}, VGPR4-7 -> +16
            const int kl = ((vg & 4) ? 16 : 0) + koffA + ((vg & 3) << 1) + w;
            a[j] = (_Float16)pooled[m * CC + kb + kl];
            // B-matrix 32x16 f16 layout: elems 0-7 -> K=koffB+{0..7}, elems 8-15 -> +16
            // B[k][n] = W[n][k]  (W row-major [d,c])
            b[j] = (_Float16)W[n * CC + kb + koffB + j + ((j & 8) ? 8 : 0)];
        }
        acc = __builtin_amdgcn_wmma_f32_16x16x32_f16(
            /*neg_a=*/false, a, /*neg_b=*/false, b,
            /*c_mod=*/(short)0, acc, /*reuse_a=*/false, /*reuse_b=*/false);
    }

    const float bias = b_conv[n] + b_gate[n];
    #pragma unroll
    for (int r = 0; r < 8; ++r) {
        // C/D 16x16 f32 layout: VGPR r -> row r (lanes 0-15) or r+8 (lanes 16-31)
        const int row   = mtile * 16 + r + ((lane < 16) ? 0 : 8);
        const float lgt = acc[r] + bias;
        g[row * CC + n] = 1.0f / (1.0f + __expf(-lgt));
    }
}

// ---------------------------------------------------------------------------
// Kernel 3: out[b,c,:,:] = sum_k g[k*B+b, c] * messages[k,b,c,:,:]
// One block per (b,c) plane; pass-2 reads should hit the 192MB L2.
// ---------------------------------------------------------------------------
__global__ void gather_kernel(const float* __restrict__ msg,
                              const float* __restrict__ g,
                              float* __restrict__ out) {
    const int bc = blockIdx.x;                   // b*C + c, 0..511
    float gv[KK];
    #pragma unroll
    for (int k = 0; k < KK; ++k) gv[k] = g[k * (BB * CC) + bc];   // uniform -> SGPRs

    float4* op = (float4*)(out + (size_t)bc * HW);
    #pragma unroll
    for (int i = 0; i < 9; ++i) {
        const int idx = threadIdx.x + (i << 8);
        float4 acc = make_float4(0.f, 0.f, 0.f, 0.f);
        #pragma unroll
        for (int k = 0; k < KK; ++k) {
            const float4* p = (const float4*)(msg + (size_t)(k * (BB * CC) + bc) * HW);
            float4 v = p[idx];
            acc.x += gv[k] * v.x;
            acc.y += gv[k] * v.y;
            acc.z += gv[k] * v.z;
            acc.w += gv[k] * v.w;
        }
        op[idx] = acc;
    }
}

// ---------------------------------------------------------------------------
extern "C" void kernel_launch(void* const* d_in, const int* in_sizes, int n_in,
                              void* d_out, int out_size, void* d_ws, size_t ws_size,
                              hipStream_t stream) {
    const float* msg    = (const float*)d_in[0];   // [K,B,C,H,W]
    const float* W      = (const float*)d_in[1];   // [C,C]
    const float* b_conv = (const float*)d_in[2];   // [C]
    const float* b_gate = (const float*)d_in[3];   // [C]
    float* out = (float*)d_out;                    // [B,C,H,W]

    float* pooled = (float*)d_ws;                  // [32,128]
    float* g      = pooled + (M_ROWS * CC);        // [32,128]

    pool_kernel<<<KK * BB * CC, 256, 0, stream>>>(msg, pooled);
    gate_kernel<<<1, 512, 0, stream>>>(pooled, W, b_conv, b_gate, g);
    gather_kernel<<<BB * CC, 256, 0, stream>>>(msg, g, out);
}
